// RNNSurrogate_34179349741726
// MI455X (gfx1250) — matrix operand order repaired and checked by reading, not compiled, not run on hardware
//
#include <hip/hip_runtime.h>

// ---------------------------------------------------------------------------
// LSTM surrogate: B=256, T=512, D=256, U=512, O=64
// Persistent-kernel design: 64 WGs x 256 threads (8 waves, wave32).
//   - WG g owns units [g*8, g*8+8) -> 32 gate columns (i,f,g,o interleave).
//   - Per step: gates tile GEMM via v_wmma_f32_16x16x32_bf16 (f32 accum)
//     with 2x2 register blocking per wave (4 indep accumulators per K-step),
//     pointwise cell update with c in LDS, h published bf16 to L2 ping-pong,
//     grid barrier, then y_t decode tile (one 16x16 tile per WG, wave 0).
// ---------------------------------------------------------------------------

#define BATCH   256
#define TSTEPS  512
#define DIN     256
#define UNITS   512
#define NOUT    64
#define GATES   2048          // 4*UNITS

#define NWG     64
#define UPW     8             // units per workgroup
#define LCOLS   32            // 4*UPW gate columns per workgroup
#define THREADS 256

typedef __attribute__((ext_vector_type(16))) __bf16 v16bf;
typedef __attribute__((ext_vector_type(8)))  float  v8f;

union FragAB {
  uint4          q[2];
  unsigned short u[16];
  v16bf          v;
};

__device__ __forceinline__ unsigned short f32_to_bf16(float f) {
  union { float f; unsigned u; } v; v.f = f;
  unsigned r = v.u + 0x7FFFu + ((v.u >> 16) & 1u);   // round-to-nearest-even
  return (unsigned short)(r >> 16);
}

__device__ __forceinline__ float sigf(float x)      { return 1.0f / (1.0f + __expf(-x)); }
__device__ __forceinline__ float tanh_fast(float x) { return 1.0f - 2.0f / (__expf(2.0f * x) + 1.0f); }

// A fragment (16x32 bf16), rows striped over lanes 0..15.
// lanes 0-15 hold K {k0..k0+7, k0+16..k0+23}; lanes 16-31 the other halves.
__device__ __forceinline__ v16bf load_a_bf16(const unsigned short* __restrict__ base,
                                             int ld, int k0, int ln, int half) {
  const unsigned short* p = base + (size_t)ln * ld + (k0 + (half << 3));
  FragAB fr;
  fr.q[0] = *(const uint4*)(p);
  fr.q[1] = *(const uint4*)(p + 16);
  return fr.v;
}

// Same A fragment but sourced from f32 (x tensor), converted on the fly.
__device__ __forceinline__ v16bf load_a_f32(const float* __restrict__ base,
                                            size_t ld, int k0, int ln, int half) {
  const float* p = base + (size_t)ln * ld + (k0 + (half << 3));
  float4 f0 = *(const float4*)(p);
  float4 f1 = *(const float4*)(p + 4);
  float4 f2 = *(const float4*)(p + 16);
  float4 f3 = *(const float4*)(p + 20);
  FragAB fr;
  fr.u[0]  = f32_to_bf16(f0.x); fr.u[1]  = f32_to_bf16(f0.y);
  fr.u[2]  = f32_to_bf16(f0.z); fr.u[3]  = f32_to_bf16(f0.w);
  fr.u[4]  = f32_to_bf16(f1.x); fr.u[5]  = f32_to_bf16(f1.y);
  fr.u[6]  = f32_to_bf16(f1.z); fr.u[7]  = f32_to_bf16(f1.w);
  fr.u[8]  = f32_to_bf16(f2.x); fr.u[9]  = f32_to_bf16(f2.y);
  fr.u[10] = f32_to_bf16(f2.z); fr.u[11] = f32_to_bf16(f2.w);
  fr.u[12] = f32_to_bf16(f3.x); fr.u[13] = f32_to_bf16(f3.y);
  fr.u[14] = f32_to_bf16(f3.z); fr.u[15] = f32_to_bf16(f3.w);
  return fr.v;
}

// B fragment (32x16 bf16) from K-major (pre-transposed) weights: wcol points
// at the start of this lane's output column; lanes 0-15 hold K {k0..k0+7, k0+16..k0+23}.
__device__ __forceinline__ v16bf load_b_kmaj(const unsigned short* __restrict__ wcol,
                                             int k0, int half) {
  const unsigned short* p = wcol + k0 + (half << 3);
  FragAB fr;
  fr.q[0] = *(const uint4*)(p);
  fr.q[1] = *(const uint4*)(p + 16);
  return fr.v;
}

__device__ __forceinline__ void grid_barrier(unsigned* __restrict__ ctr) {
  __builtin_amdgcn_fence(__ATOMIC_RELEASE, "agent");   // publish h stores to L2
  __syncthreads();
  if (threadIdx.x == 0) {
    __hip_atomic_fetch_add(ctr, 1u, __ATOMIC_ACQ_REL, __HIP_MEMORY_SCOPE_AGENT);
    while (__hip_atomic_load(ctr, __ATOMIC_ACQUIRE, __HIP_MEMORY_SCOPE_AGENT) < NWG) {
      __builtin_amdgcn_s_sleep(1);
    }
  }
  __syncthreads();
  __builtin_amdgcn_fence(__ATOMIC_ACQUIRE, "agent");   // invalidate WGP$ for h reads
}

// ---------------------------------------------------------------------------
// Init kernel: build K-major bf16 weight copies, zero h ping-pong + barriers.
// Runs every launch (stream-ordered before the persistent kernel).
// ---------------------------------------------------------------------------
__global__ void lstm_init(const float* __restrict__ Wx, const float* __restrict__ Wh,
                          const float* __restrict__ Wd,
                          unsigned short* __restrict__ WxT, unsigned short* __restrict__ WhT,
                          unsigned short* __restrict__ WdT, unsigned short* __restrict__ Hbuf,
                          unsigned* __restrict__ bar) {
  const long NWX = (long)GATES * DIN;
  const long NWH = (long)GATES * UNITS;
  const long NWD = (long)NOUT * UNITS;
  const long NH  = 2L * BATCH * UNITS;
  const long NB  = TSTEPS;
  const long total = NWX + NWH + NWD + NH + NB;
  for (long i = blockIdx.x * (long)blockDim.x + threadIdx.x; i < total;
       i += (long)gridDim.x * blockDim.x) {
    long j = i;
    if (j < NWX) {                       // WxT[gcol][k] = Wx[k][gcol]
      long gc = j / DIN, k = j % DIN;
      WxT[j] = f32_to_bf16(Wx[k * GATES + gc]);
      continue;
    }
    j -= NWX;
    if (j < NWH) {                       // WhT[gcol][k] = Wh[k][gcol]
      long gc = j / UNITS, k = j % UNITS;
      WhT[j] = f32_to_bf16(Wh[k * GATES + gc]);
      continue;
    }
    j -= NWH;
    if (j < NWD) {                       // WdT[o][k] = Wd[k][o]
      long o = j / UNITS, k = j % UNITS;
      WdT[j] = f32_to_bf16(Wd[k * NOUT + o]);
      continue;
    }
    j -= NWD;
    if (j < NH) { Hbuf[j] = 0; continue; }   // h0 = 0 (both slots)
    j -= NH;
    bar[j] = 0u;                              // per-step barrier counters
  }
}

// ---------------------------------------------------------------------------
// Persistent LSTM kernel.
// ---------------------------------------------------------------------------
__global__ void __launch_bounds__(THREADS, 1)
lstm_persistent(const float* __restrict__ x, const float* __restrict__ bias,
                const float* __restrict__ bd,
                const unsigned short* __restrict__ WxT,
                const unsigned short* __restrict__ WhT,
                const unsigned short* __restrict__ WdT,
                unsigned short* __restrict__ Hbuf, unsigned* __restrict__ bar,
                float* __restrict__ y) {
  __shared__ float gatesLds[BATCH * LCOLS];   // 32 KB: pre-activation gate slice
  __shared__ float cLds[BATCH * UPW];         //  8 KB: cell state slice (WG-private)

  const int wg   = blockIdx.x;
  const int u0   = wg * UPW;
  const int lane = threadIdx.x & 31;
  const int wave = threadIdx.x >> 5;
  const int ln   = lane & 15;
  const int half = lane >> 4;

  for (int i = threadIdx.x; i < BATCH * UPW; i += THREADS) cLds[i] = 0.0f;
  __syncthreads();

  const size_t xld = (size_t)TSTEPS * DIN;    // batch-row stride of x

  // Per-lane B column pointers for the wave's two N-tiles (fixed per WG).
  const int lcol0 = ln;                        // N-tile 0 -> gates i,f
  const int lcol1 = 16 + ln;                   // N-tile 1 -> gates g,o
  const int gcol0 = (lcol0 >> 3) * UNITS + u0 + (lcol0 & 7);
  const int gcol1 = (lcol1 >> 3) * UNITS + u0 + (lcol1 & 7);
  const unsigned short* wx0 = WxT + (size_t)gcol0 * DIN;
  const unsigned short* wx1 = WxT + (size_t)gcol1 * DIN;
  const unsigned short* wh0 = WhT + (size_t)gcol0 * UNITS;
  const unsigned short* wh1 = WhT + (size_t)gcol1 * UNITS;

  for (int t = 0; t < TSTEPS; ++t) {
    const int cur = t & 1;
    const unsigned short* Hcur = Hbuf + (size_t)cur * BATCH * UNITS;
    unsigned short*       Hnxt = Hbuf + (size_t)(cur ^ 1) * BATCH * UNITS;

    // ---- Phase 1: gate GEMM  G[256 x 32] = x_t @ Wx_slice + h @ Wh_slice ----
    // Wave owns rows [wave*32, wave*32+32): 2 M-tiles x 2 N-tiles register
    // block -> per K-step: 2 A-frag + 2 B-frag loads feed 4 WMMAs.
    {
      const int mrow0 = wave * 32;
      const float*          xb0 = x + (size_t)mrow0 * xld + (size_t)t * DIN;
      const float*          xb1 = xb0 + (size_t)16 * xld;
      const unsigned short* hb0 = Hcur + (size_t)mrow0 * UNITS;
      const unsigned short* hb1 = hb0 + (size_t)16 * UNITS;

      v8f acc00 = {0.f, 0.f, 0.f, 0.f, 0.f, 0.f, 0.f, 0.f};
      v8f acc01 = acc00, acc10 = acc00, acc11 = acc00;

#pragma unroll 2
      for (int k0 = 0; k0 < DIN; k0 += 32) {
        const v16bf a0 = load_a_f32(xb0, xld, k0, ln, half);
        const v16bf a1 = load_a_f32(xb1, xld, k0, ln, half);
        const v16bf b0 = load_b_kmaj(wx0, k0, half);
        const v16bf b1 = load_b_kmaj(wx1, k0, half);
        acc00 = __builtin_amdgcn_wmma_f32_16x16x32_bf16(false, a0, false, b0, (short)0, acc00, false, false);
        acc01 = __builtin_amdgcn_wmma_f32_16x16x32_bf16(false, a0, false, b1, (short)0, acc01, false, false);
        acc10 = __builtin_amdgcn_wmma_f32_16x16x32_bf16(false, a1, false, b0, (short)0, acc10, false, false);
        acc11 = __builtin_amdgcn_wmma_f32_16x16x32_bf16(false, a1, false, b1, (short)0, acc11, false, false);
      }
#pragma unroll 2
      for (int k0 = 0; k0 < UNITS; k0 += 32) {
        const v16bf a0 = load_a_bf16(hb0, UNITS, k0, ln, half);
        const v16bf a1 = load_a_bf16(hb1, UNITS, k0, ln, half);
        const v16bf b0 = load_b_kmaj(wh0, k0, half);
        const v16bf b1 = load_b_kmaj(wh1, k0, half);
        acc00 = __builtin_amdgcn_wmma_f32_16x16x32_bf16(false, a0, false, b0, (short)0, acc00, false, false);
        acc01 = __builtin_amdgcn_wmma_f32_16x16x32_bf16(false, a0, false, b1, (short)0, acc01, false, false);
        acc10 = __builtin_amdgcn_wmma_f32_16x16x32_bf16(false, a1, false, b0, (short)0, acc10, false, false);
        acc11 = __builtin_amdgcn_wmma_f32_16x16x32_bf16(false, a1, false, b1, (short)0, acc11, false, false);
      }

      // C-fragment layout: element (M = v + 8*half, N = ln)
      const int r0 = mrow0 + (half << 3);
#pragma unroll
      for (int v = 0; v < 8; ++v) {
        gatesLds[(r0 + v) * LCOLS + ln]           = acc00[v];
        gatesLds[(r0 + v) * LCOLS + 16 + ln]      = acc01[v];
        gatesLds[(r0 + 16 + v) * LCOLS + ln]      = acc10[v];
        gatesLds[(r0 + 16 + v) * LCOLS + 16 + ln] = acc11[v];
      }
    }
    __syncthreads();

    // ---- Phase 2: cell update (thread = batch row; 8 units each) ----
    {
      const int row = threadIdx.x;
      const float*    grow = gatesLds + row * LCOLS;
      float*          crow = cLds + row * UPW;
      unsigned short* hrow = Hnxt + (size_t)row * UNITS + u0;
#pragma unroll
      for (int uu = 0; uu < UPW; ++uu) {
        const float iv = grow[uu]      + bias[u0 + uu];
        const float fv = grow[8 + uu]  + bias[UNITS + u0 + uu];
        const float gv = grow[16 + uu] + bias[2 * UNITS + u0 + uu];
        const float ov = grow[24 + uu] + bias[3 * UNITS + u0 + uu];
        const float cN = sigf(fv) * crow[uu] + sigf(iv) * tanh_fast(gv);
        const float h  = sigf(ov) * tanh_fast(cN);
        crow[uu] = cN;
        hrow[uu] = f32_to_bf16(h);
      }
    }

    // ---- Phase 3: device-wide barrier (h_{t+1} published) ----
    grid_barrier(&bar[t]);

    // ---- Phase 4: decode y_t = h_{t+1} @ Wd + bd ----
    // 64 tiles (16 M x 4 N) == 64 WGs: one 16x16 tile on wave 0 of each WG,
    // overlapped with the next step's GEMM on waves 1..7.
    if (wave == 0) {
      const int mt = wg >> 2, ot = wg & 3;
      const int mrow0 = mt * 16;
      const int oc = ot * 16 + ln;
      const unsigned short* hbase = Hnxt + (size_t)mrow0 * UNITS;
      const unsigned short* wdc   = WdT + (size_t)oc * UNITS;
      v8f acc = {0.f, 0.f, 0.f, 0.f, 0.f, 0.f, 0.f, 0.f};
#pragma unroll 4
      for (int k0 = 0; k0 < UNITS; k0 += 32) {
        const v16bf a = load_a_bf16(hbase, UNITS, k0, ln, half);
        const v16bf b = load_b_kmaj(wdc, k0, half);
        acc = __builtin_amdgcn_wmma_f32_16x16x32_bf16(
            false, a, false, b, (short)0, acc, false, false);
      }
      const float bo = bd[oc];
#pragma unroll
      for (int v = 0; v < 8; ++v) {
        const int brow = mrow0 + v + (half << 3);
        y[((size_t)brow * TSTEPS + t) * NOUT + oc] = acc[v] + bo;
      }
    }
  }
}

// ---------------------------------------------------------------------------
extern "C" void kernel_launch(void* const* d_in, const int* in_sizes, int n_in,
                              void* d_out, int out_size, void* d_ws, size_t ws_size,
                              hipStream_t stream) {
  (void)in_sizes; (void)n_in; (void)out_size; (void)ws_size;
  const float* x    = (const float*)d_in[0];   // [B,T,D]
  const float* Wx   = (const float*)d_in[1];   // [D,4U]
  const float* Wh   = (const float*)d_in[2];   // [U,4U]
  const float* bias = (const float*)d_in[3];   // [4U]
  const float* Wd   = (const float*)d_in[4];   // [U,O]
  const float* bd   = (const float*)d_in[5];   // [O]
  float* y = (float*)d_out;                    // [B,T,O]

  // Workspace layout (all sizes multiples of 256B): ~3.7 MB total.
  char* p = (char*)d_ws;
  unsigned short* WxT  = (unsigned short*)p; p += (size_t)GATES * DIN * 2;
  unsigned short* WhT  = (unsigned short*)p; p += (size_t)GATES * UNITS * 2;
  unsigned short* WdT  = (unsigned short*)p; p += (size_t)NOUT * UNITS * 2;
  unsigned short* Hbuf = (unsigned short*)p; p += (size_t)2 * BATCH * UNITS * 2;
  unsigned*       bar  = (unsigned*)p;       p += (size_t)TSTEPS * 4;

  lstm_init<<<1024, 256, 0, stream>>>(Wx, Wh, Wd, WxT, WhT, WdT, Hbuf, bar);
  lstm_persistent<<<NWG, THREADS, 0, stream>>>(x, bias, bd, WxT, WhT, WdT, Hbuf, bar, y);
}
